// _QuantumImageEncoder_65481071409405
// MI455X (gfx1250) — matrix operand order repaired and checked by reading, not compiled, not run on hardware
//
#include <hip/hip_runtime.h>
#include <hip/hip_bf16.h>

typedef __attribute__((ext_vector_type(16))) _Float16 v16h;
typedef __attribute__((ext_vector_type(8)))  _Float16 v8h;
typedef __attribute__((ext_vector_type(8)))  float    v8f;

// ---------------------------------------------------------------------------
// Kernel 0: fc1_w [64,784] f32 -> f16 [64,800] (K padded with zeros to 800)
// ---------------------------------------------------------------------------
__global__ void k_prep_w1(const float* __restrict__ w, _Float16* __restrict__ wh) {
    int idx = blockIdx.x * blockDim.x + threadIdx.x;
    if (idx >= 64 * 800) return;
    int n = idx / 800, k = idx % 800;
    float v = (k < 784) ? w[n * 784 + k] : 0.0f;
    wh[idx] = (_Float16)v;
}

// ---------------------------------------------------------------------------
// Kernel 1: per-image fused pipeline.
//   conv1 (1->8, 3x3 SAME) + relu + 2x2 pool  : VALU (0.9 GF total)
//   conv2 (8->16, 3x3 SAME)                   : implicit-GEMM WMMA
//       A: h1 stored channel-minor f16 with zero halo -> one 16B ds_load per
//          (position, shift) gives 8 contiguous K values. K = shift*8+ic,
//          9 shifts padded to 12 -> K=96 = 3 x 32.
//       B: weights reordered to [n][96] f16 in LDS, padded shifts zeroed.
//   relu + 2x2 pool + f16 store to flat [B,800] (cols 784..799 zero).
// ---------------------------------------------------------------------------
__global__ void k_conv(const float* __restrict__ x,
                       const float* __restrict__ w1, const float* __restrict__ b1,
                       const float* __restrict__ w2, const float* __restrict__ b2,
                       _Float16* __restrict__ flat) {
    __shared__ float    img[784];          // 28x28 input
    __shared__ float    wc1[72];
    __shared__ float    bc1[8];
    __shared__ _Float16 h1c[18][16][8];    // conv1 pooled, channel-minor, halo
    __shared__ _Float16 wh2[16 * 96];      // conv2 weights [n][k], k=shift*8+ic
    __shared__ float    bc2[16];
    __shared__ float    c2[208 * 16];      // conv2 output [pos][channel]

    const int tid = threadIdx.x;
    const int b   = blockIdx.x;
    const float* xim = x + (size_t)b * 784;

    // ---- stage inputs / weights; zero h1c halo ----
    for (int i = tid; i < 784; i += 256) img[i] = xim[i];
    for (int i = tid; i < 72;  i += 256) wc1[i] = w1[i];
    if (tid < 8)  bc1[tid] = b1[tid];
    if (tid < 16) bc2[tid] = b2[tid];
    {
        _Float16* h1f = &h1c[0][0][0];
        for (int i = tid; i < 18 * 16 * 8; i += 256) h1f[i] = (_Float16)0.0f;
    }
    for (int i = tid; i < 16 * 96; i += 256) {
        int n = i / 96, k = i % 96;
        int shift = k / 8, ic = k % 8;
        float v = (shift < 9) ? w2[n * 72 + ic * 9 + shift] : 0.0f;
        wh2[i] = (_Float16)v;
    }
    __syncthreads();

    // ---- conv1 + relu + pool -> h1c[py+1][px+1][c] ----
    for (int o = tid; o < 1568; o += 256) {
        int c = o / 196, r = o % 196, py = r / 14, px = r % 14;
        float mx = -1e30f;
        for (int sy = 0; sy < 2; ++sy)
        for (int sx = 0; sx < 2; ++sx) {
            int Y = 2 * py + sy, X = 2 * px + sx;
            float acc = bc1[c];
            #pragma unroll
            for (int dy = 0; dy < 3; ++dy) {
                int iy = Y + dy - 1;
                if (iy < 0 || iy >= 28) continue;
                #pragma unroll
                for (int dx = 0; dx < 3; ++dx) {
                    int ix = X + dx - 1;
                    if (ix < 0 || ix >= 28) continue;
                    acc += img[iy * 28 + ix] * wc1[c * 9 + dy * 3 + dx];
                }
            }
            mx = fmaxf(mx, acc);
        }
        h1c[py + 1][px + 1][c] = (_Float16)fmaxf(mx, 0.0f);
    }
    __syncthreads();

    // ---- conv2 as implicit-GEMM WMMA: M=196(->208), N=16, K=96 ----
    const int lane = tid & 31;
    const int wv   = tid >> 5;
    const int half = lane >> 4;
    const int lid  = lane & 15;
    const float biasn = bc2[lid];
    for (int t = wv; t < 13; t += 8) {           // 13 M-tiles of 16 positions
        const int m0t = t * 16;
        const int m = m0t + lid;                 // this lane's A row (position)
        const int y = m / 14, xp = m % 14;       // garbage rows m>=196 read halo
        v8f acc = {};
        #pragma unroll
        for (int s = 0; s < 3; ++s) {            // K steps of 32 = 4 shifts x 8 ic
            const int sh0 = 4 * s + 2 * half;    // shifts for elements 0..7
            const int sh1 = 4 * s + 2 * half + 1; // shifts for elements 8..15
            v8h alo = *(const v8h*)&h1c[y + sh0 / 3][xp + sh0 % 3][0];
            v8h ahi = *(const v8h*)&h1c[y + sh1 / 3][xp + sh1 % 3][0];
            v16h a;
            #pragma unroll
            for (int i = 0; i < 8; ++i) { a[i] = alo[i]; a[i + 8] = ahi[i]; }
            v16h bf = *(const v16h*)&wh2[lid * 96 + s * 32 + half * 16];
            acc = __builtin_amdgcn_wmma_f32_16x16x32_f16(false, a, false, bf,
                                                         (short)0, acc, false, false);
        }
        #pragma unroll
        for (int r = 0; r < 8; ++r) {            // D[M=r+8*half][N=lid]
            int M = m0t + r + 8 * half;
            c2[M * 16 + lid] = fmaxf(acc[r] + biasn, 0.0f);
        }
    }
    __syncthreads();

    // ---- 2x2 maxpool over c2 positions -> flat f16 [c*49 + y7*7 + x7] ----
    _Float16* frow = flat + (size_t)b * 800;
    for (int o = tid; o < 784; o += 256) {
        int c = o / 49, r = o % 49, y7 = r / 7, x7 = r % 7;
        float v00 = c2[((2 * y7)     * 14 + 2 * x7)     * 16 + c];
        float v01 = c2[((2 * y7)     * 14 + 2 * x7 + 1) * 16 + c];
        float v10 = c2[((2 * y7 + 1) * 14 + 2 * x7)     * 16 + c];
        float v11 = c2[((2 * y7 + 1) * 14 + 2 * x7 + 1) * 16 + c];
        frow[o] = (_Float16)fmaxf(fmaxf(v00, v01), fmaxf(v10, v11));
    }
    if (tid < 16) frow[784 + tid] = (_Float16)0.0f;   // K padding
}

// ---------------------------------------------------------------------------
// Kernel 2: fc1 (784->64) via v_wmma_f32_16x16x32_f16, relu, then fc2 (64->4).
// 8 waves/block; each wave computes a 16(M) x 64(N) tile, K-loop of 25 x 32.
// ---------------------------------------------------------------------------
__global__ void k_fc_wmma(const _Float16* __restrict__ flat,
                          const _Float16* __restrict__ w1h,
                          const float* __restrict__ fb1,
                          const float* __restrict__ w2,
                          const float* __restrict__ fb2,
                          float* __restrict__ feat, int B) {
    __shared__ float hsh[8][16][64];
    const int lane = threadIdx.x & 31;
    const int wv   = threadIdx.x >> 5;
    const int m0   = (blockIdx.x * 8 + wv) * 16;
    if (m0 >= B) return;                         // uniform per wave
    const int half = lane >> 4;                  // 0: lanes 0-15, 1: 16-31
    const int lid  = lane & 15;

    v8f acc0 = {}, acc1 = {}, acc2 = {}, acc3 = {};
    const _Float16* arow = flat + (size_t)(m0 + lid) * 800 + half * 16;
    const _Float16* bp0 = w1h + (size_t)( 0 + lid) * 800 + half * 16;
    const _Float16* bp1 = w1h + (size_t)(16 + lid) * 800 + half * 16;
    const _Float16* bp2 = w1h + (size_t)(32 + lid) * 800 + half * 16;
    const _Float16* bp3 = w1h + (size_t)(48 + lid) * 800 + half * 16;

    for (int k0 = 0; k0 < 800; k0 += 32) {
        __builtin_prefetch(arow + k0 + 64, 0, 1);
        v8h alo = *(const v8h*)(arow + k0);
        v8h ahi = *(const v8h*)(arow + k0 + 8);
        v16h a;
        #pragma unroll
        for (int i = 0; i < 8; ++i) { a[i] = alo[i]; a[i + 8] = ahi[i]; }
        v16h b0 = *(const v16h*)(bp0 + k0);
        v16h b1 = *(const v16h*)(bp1 + k0);
        v16h b2 = *(const v16h*)(bp2 + k0);
        v16h b3 = *(const v16h*)(bp3 + k0);
        acc0 = __builtin_amdgcn_wmma_f32_16x16x32_f16(false, a, false, b0, (short)0, acc0, false, false);
        acc1 = __builtin_amdgcn_wmma_f32_16x16x32_f16(false, a, false, b1, (short)0, acc1, false, false);
        acc2 = __builtin_amdgcn_wmma_f32_16x16x32_f16(false, a, false, b2, (short)0, acc2, false, false);
        acc3 = __builtin_amdgcn_wmma_f32_16x16x32_f16(false, a, false, b3, (short)0, acc3, false, false);
    }

    #pragma unroll
    for (int r = 0; r < 8; ++r) {
        int M = r + 8 * half;
        hsh[wv][M][ 0 + lid] = fmaxf(acc0[r] + fb1[ 0 + lid], 0.0f);
        hsh[wv][M][16 + lid] = fmaxf(acc1[r] + fb1[16 + lid], 0.0f);
        hsh[wv][M][32 + lid] = fmaxf(acc2[r] + fb1[32 + lid], 0.0f);
        hsh[wv][M][48 + lid] = fmaxf(acc3[r] + fb1[48 + lid], 0.0f);
    }
    __syncthreads();

    // fc2: lane handles sample m=lid, outputs j=half and j=half+2
    int j = half;
    float a0 = fb2[j], a1 = fb2[j + 2];
    #pragma unroll 8
    for (int n = 0; n < 64; ++n) {
        float hv = hsh[wv][lid][n];
        a0 += hv * w2[j * 64 + n];
        a1 += hv * w2[(j + 2) * 64 + n];
    }
    feat[(size_t)(m0 + lid) * 4 + j]     = a0;
    feat[(size_t)(m0 + lid) * 4 + j + 2] = a1;
}

// ---------------------------------------------------------------------------
// Kernel 3: batch-norm stats (biased var, deterministic tree reduce) folded
// into scale/shift; precompute cos/sin of the 23 shared gate params.
// ---------------------------------------------------------------------------
__global__ void k_stats(const float* __restrict__ feat,
                        const float* __restrict__ bng, const float* __restrict__ bnb,
                        const float* __restrict__ rl,
                        float* __restrict__ bnp, float* __restrict__ trig, int B) {
    __shared__ float s_sum[256], s_sq[256];
    const int tid = threadIdx.x;
    const int f = tid >> 6, s0 = tid & 63;
    float sum = 0.0f, sq = 0.0f;
    for (int i = s0; i < B; i += 64) {
        float v = feat[(size_t)i * 4 + f];
        sum += v; sq += v * v;
    }
    s_sum[tid] = sum; s_sq[tid] = sq;
    __syncthreads();
    for (int st = 32; st > 0; st >>= 1) {
        if (s0 < st) { s_sum[tid] += s_sum[tid + st]; s_sq[tid] += s_sq[tid + st]; }
        __syncthreads();
    }
    if (s0 == 0) {
        float invB = 1.0f / (float)B;
        float mean = s_sum[tid] * invB;
        float var  = s_sq[tid] * invB - mean * mean;
        float sc   = bng[f] * rsqrtf(var + 1e-5f);
        bnp[f]     = sc;
        bnp[4 + f] = bnb[f] - mean * sc;
    }
    if (tid < 23) {
        float p = rl[tid] * 0.5f;
        trig[2 * tid]     = cosf(p);
        trig[2 * tid + 1] = sinf(p);
    }
}

// ---------------------------------------------------------------------------
// Kernel 4: 4-qubit state-vector sim, one sample per lane, fully unrolled.
// Wire w <-> bit (3-w): masks w0=8, w1=4, w2=2, w3=1.
// ---------------------------------------------------------------------------
template<int MASK>
__device__ __forceinline__ void g_rx(float* sr, float* si, float c, float s) {
    #pragma unroll
    for (int i = 0; i < 16; ++i) if (!(i & MASK)) {
        int j = i | MASK;
        float ar = sr[i], ai = si[i], br = sr[j], bi = si[j];
        sr[i] = c * ar + s * bi;  si[i] = c * ai - s * br;
        sr[j] = c * br + s * ai;  si[j] = c * bi - s * ar;
    }
}
template<int MASK>
__device__ __forceinline__ void g_ry(float* sr, float* si, float c, float s) {
    #pragma unroll
    for (int i = 0; i < 16; ++i) if (!(i & MASK)) {
        int j = i | MASK;
        float ar = sr[i], ai = si[i], br = sr[j], bi = si[j];
        sr[i] = c * ar - s * br;  si[i] = c * ai - s * bi;
        sr[j] = s * ar + c * br;  si[j] = s * ai + c * bi;
    }
}
template<int MASK>
__device__ __forceinline__ void g_rz(float* sr, float* si, float c, float s) {
    #pragma unroll
    for (int i = 0; i < 16; ++i) if (!(i & MASK)) {
        int j = i | MASK;
        float ar = sr[i], ai = si[i], br = sr[j], bi = si[j];
        sr[i] = c * ar + s * ai;  si[i] = c * ai - s * ar;
        sr[j] = c * br - s * bi;  si[j] = c * bi + s * br;
    }
}

__global__ void k_quantum(const float* __restrict__ feat,
                          const float* __restrict__ bnp,
                          const float* __restrict__ trig,
                          float* __restrict__ out, int B) {
    int b = blockIdx.x * blockDim.x + threadIdx.x;
    if (b >= B) return;
    float sr[16], si[16];
    #pragma unroll
    for (int i = 0; i < 16; ++i) { sr[i] = 0.0f; si[i] = 0.0f; }
    sr[0] = 1.0f;

    float f0 = feat[(size_t)b * 4 + 0] * bnp[0] + bnp[4];
    float f1 = feat[(size_t)b * 4 + 1] * bnp[1] + bnp[5];
    float f2 = feat[(size_t)b * 4 + 2] * bnp[2] + bnp[6];
    float f3 = feat[(size_t)b * 4 + 3] * bnp[3] + bnp[7];
    g_rx<8>(sr, si, cosf(0.5f * f0), sinf(0.5f * f0));
    g_rx<4>(sr, si, cosf(0.5f * f1), sinf(0.5f * f1));
    g_rx<2>(sr, si, cosf(0.5f * f2), sinf(0.5f * f2));
    g_rx<1>(sr, si, cosf(0.5f * f3), sinf(0.5f * f3));

    #pragma unroll
    for (int g = 0; g < 7; ++g) {
        g_rx<8>(sr, si, trig[6 * g + 0], trig[6 * g + 1]);
        g_ry<4>(sr, si, trig[6 * g + 2], trig[6 * g + 3]);
        g_rz<2>(sr, si, trig[6 * g + 4], trig[6 * g + 5]);
        #pragma unroll
        for (int i = 1; i < 8; i += 2) {      // CNOT(ctrl=w3 bit0, tgt=w0 bit3)
            float t;
            t = sr[i]; sr[i] = sr[i | 8]; sr[i | 8] = t;
            t = si[i]; si[i] = si[i | 8]; si[i | 8] = t;
        }
    }
    g_rx<8>(sr, si, trig[42], trig[43]);
    g_ry<4>(sr, si, trig[44], trig[45]);

    float e0 = 0, e1 = 0, e2 = 0, e3 = 0;
    #pragma unroll
    for (int i = 0; i < 16; ++i) {
        float p = sr[i] * sr[i] + si[i] * si[i];
        e0 += (i & 8) ? -p : p;
        e1 += (i & 4) ? -p : p;
        e2 += (i & 2) ? -p : p;
        e3 += (i & 1) ? -p : p;
    }
    out[(size_t)b * 4 + 0] = e0;
    out[(size_t)b * 4 + 1] = e1;
    out[(size_t)b * 4 + 2] = e2;
    out[(size_t)b * 4 + 3] = e3;
}

// ---------------------------------------------------------------------------
extern "C" void kernel_launch(void* const* d_in, const int* in_sizes, int n_in,
                              void* d_out, int out_size, void* d_ws, size_t ws_size,
                              hipStream_t stream) {
    const float* x   = (const float*)d_in[0];
    const float* c1w = (const float*)d_in[1];
    const float* c1b = (const float*)d_in[2];
    const float* c2w = (const float*)d_in[3];
    const float* c2b = (const float*)d_in[4];
    const float* f1w = (const float*)d_in[5];
    const float* f1b = (const float*)d_in[6];
    const float* f2w = (const float*)d_in[7];
    const float* f2b = (const float*)d_in[8];
    const float* bng = (const float*)d_in[9];
    const float* bnb = (const float*)d_in[10];
    const float* rlp = (const float*)d_in[11];
    const int B = in_sizes[0] / 784;          // 8192

    char* ws = (char*)d_ws;
    size_t off = 0;
    _Float16* flat = (_Float16*)(ws + off); off += (size_t)B * 800 * sizeof(_Float16);
    _Float16* w1h  = (_Float16*)(ws + off); off += (size_t)64 * 800 * sizeof(_Float16);
    float*    feat = (float*)(ws + off);    off += (size_t)B * 4 * sizeof(float);
    float*    bnp  = (float*)(ws + off);    off += 8 * sizeof(float);
    float*    trig = (float*)(ws + off);    off += 46 * sizeof(float);
    float*    out  = (float*)d_out;

    k_prep_w1<<<(64 * 800 + 255) / 256, 256, 0, stream>>>(f1w, w1h);
    k_conv<<<B, 256, 0, stream>>>(x, c1w, c1b, c2w, c2b, flat);
    k_fc_wmma<<<(B / 16 + 7) / 8, 256, 0, stream>>>(flat, w1h, f1b, f2w, f2b, feat, B);
    k_stats<<<1, 256, 0, stream>>>(feat, bng, bnb, rlp, bnp, trig, B);
    k_quantum<<<(B + 255) / 256, 256, 0, stream>>>(feat, bnp, trig, out, B);
}
